// PrototypeBase_20349555048831
// MI455X (gfx1250) — matrix operand order repaired and checked by reading, not compiled, not run on hardware
//
#include <hip/hip_runtime.h>
#include <hip/hip_bf16.h>
#include <math.h>

#define N_ROWS 262144
#define DIM    256
#define NPROTO 512
#define REG1C  0.05f
#define REG2C  0.05f

typedef __attribute__((ext_vector_type(16))) __bf16 v16bf;
typedef __attribute__((ext_vector_type(8)))  float  v8f;

union Frag {
    unsigned u[8];
    uint4    q[2];
    v16bf    v;
};

__device__ __forceinline__ unsigned short f32_to_bf16_rne(float x) {
    unsigned u = __float_as_uint(x);
    unsigned r = (u + 0x7FFFu + ((u >> 16) & 1u)) >> 16;
    return (unsigned short)r;
}
__device__ __forceinline__ float bf16u_to_f32(unsigned short h) {
    return __uint_as_float(((unsigned)h) << 16);
}

// ---------------------------------------------------------------------------
__global__ void init_kernel(unsigned* __restrict__ colmin, float* __restrict__ out) {
    int t = threadIdx.x;
    if (t < NPROTO) colmin[t] = 0x7F800000u;   // +inf bits
    if (t == 0) out[0] = 0.0f;
}

// Convert prototypes to hi/lo bf16 split, compute |p|^2 per prototype.
__global__ void proto_prep(const float* __restrict__ proto,
                           unsigned short* __restrict__ ph,
                           unsigned short* __restrict__ pl,
                           float* __restrict__ p2) {
    int p = blockIdx.x * blockDim.x + threadIdx.x;
    if (p >= NPROTO) return;
    float s = 0.0f;
    const float* row = proto + (size_t)p * DIM;
    unsigned short* hrow = ph + (size_t)p * DIM;
    unsigned short* lrow = pl + (size_t)p * DIM;
    for (int d = 0; d < DIM; ++d) {
        float x = row[d];
        unsigned short hi = f32_to_bf16_rne(x);
        unsigned short lo = f32_to_bf16_rne(x - bf16u_to_f32(hi));
        hrow[d] = hi;
        lrow[d] = lo;
        s = fmaf(x, x, s);
    }
    p2[p] = s;
}

// ---------------------------------------------------------------------------
// Main distance kernel: block = 8 wave32s, 128 rows of z; each wave owns a
// 16-row strip and sweeps all 512 prototype columns with split-bf16 WMMA.
__global__ __launch_bounds__(256)
void proto_dist_kernel(const float* __restrict__ z,
                       const unsigned short* __restrict__ ph,
                       const unsigned short* __restrict__ pl,
                       const float* __restrict__ p2,
                       unsigned* __restrict__ colmin_g,
                       float* __restrict__ out) {
    __shared__ unsigned colmin_s[NPROTO];
    __shared__ float    rowacc;

    const int tid   = threadIdx.x;
    const int wave  = tid >> 5;
    const int lane  = tid & 31;
    const int laneN = lane & 15;
    const int half  = lane >> 4;
    const int khalf = half * 8;
    const int rowA  = blockIdx.x * 128 + wave * 16;

    for (int c = tid; c < NPROTO; c += 256) colmin_s[c] = 0x7F800000u;
    if (tid == 0) rowacc = 0.0f;
    __syncthreads();

    // --- Build register-resident A fragments (ISA 16-bit A layout):
    // lane holds row m = rowA+laneN; per K-tile two contiguous runs of 8
    // elements at K = kt*32 + 8*half and +16 more.
    Frag Ah[8], Al[8];
    float z2part = 0.0f;
    const float* zrow = z + (size_t)(rowA + laneN) * DIM;
    #pragma unroll
    for (int kt = 0; kt < 8; ++kt) {
        #pragma unroll
        for (int c = 0; c < 2; ++c) {
            const float* src = zrow + kt * 32 + khalf + c * 16;
            #pragma unroll
            for (int p = 0; p < 4; ++p) {
                float2 xx = *(const float2*)(src + 2 * p);
                unsigned short h0 = f32_to_bf16_rne(xx.x);
                unsigned short h1 = f32_to_bf16_rne(xx.y);
                unsigned short l0 = f32_to_bf16_rne(xx.x - bf16u_to_f32(h0));
                unsigned short l1 = f32_to_bf16_rne(xx.y - bf16u_to_f32(h1));
                Ah[kt].u[c * 4 + p] = (unsigned)h0 | ((unsigned)h1 << 16);
                Al[kt].u[c * 4 + p] = (unsigned)l0 | ((unsigned)l1 << 16);
                z2part = fmaf(xx.x, xx.x, z2part);
                z2part = fmaf(xx.y, xx.y, z2part);
            }
        }
    }
    // Full |z_m|^2 for row m = rowA+laneN (two lane-halves hold disjoint K).
    float z2row = z2part + __shfl_xor(z2part, 16, 32);
    // Redistribute to C-layout rows: acc element i covers row i + 8*half.
    float z2c[8];
    #pragma unroll
    for (int i = 0; i < 8; ++i) z2c[i] = __shfl(z2row, i + khalf, 32);

    float rm[8];
    #pragma unroll
    for (int i = 0; i < 8; ++i) rm[i] = 3.4e38f;

    for (int ct = 0; ct < 32; ++ct) {
        const int pr = ct * 16 + laneN;            // prototype (B column) index
        const unsigned short* phr = ph + (size_t)pr * DIM;
        const unsigned short* plr = pl + (size_t)pr * DIM;
        if (pr + 16 < NPROTO) {
            __builtin_prefetch(phr + DIM * 16, 0, 3);
            __builtin_prefetch(plr + DIM * 16, 0, 3);
        }
        const float p2v = p2[pr];

        v8f acc = {};
        #pragma unroll
        for (int kt = 0; kt < 8; ++kt) {
            // B layout: lane half holds K runs 8*half..+7 and 16+8*half..+7 -> two b128 loads.
            Frag Bh, Bl;
            const unsigned short* bh = phr + kt * 32 + half * 8;
            const unsigned short* bl = plr + kt * 32 + half * 8;
            Bh.q[0] = *(const uint4*)(bh);
            Bh.q[1] = *(const uint4*)(bh + 16);
            Bl.q[0] = *(const uint4*)(bl);
            Bl.q[1] = *(const uint4*)(bl + 16);

            acc = __builtin_amdgcn_wmma_f32_16x16x32_bf16(false, Ah[kt].v, false, Bh.v,
                                                          (short)0, acc, false, false);
            acc = __builtin_amdgcn_wmma_f32_16x16x32_bf16(false, Al[kt].v, false, Bh.v,
                                                          (short)0, acc, false, false);
            acc = __builtin_amdgcn_wmma_f32_16x16x32_bf16(false, Ah[kt].v, false, Bl.v,
                                                          (short)0, acc, false, false);
        }

        float cmin = 3.4e38f;
        #pragma unroll
        for (int i = 0; i < 8; ++i) {
            float sq = z2c[i] + p2v - 2.0f * acc[i];
            sq = fmaxf(sq, 0.0f);
            rm[i] = fminf(rm[i], sq);
            cmin  = fminf(cmin, sq);
        }
        cmin = fminf(cmin, __shfl_xor(cmin, 16, 32));  // min over the wave's 16 rows
        atomicMin(&colmin_s[ct * 16 + laneN], __float_as_uint(cmin));
    }

    // Row-min reduction across the 16 lanes of each half (masks keep bit4).
    #pragma unroll
    for (int m = 1; m <= 8; m <<= 1) {
        #pragma unroll
        for (int i = 0; i < 8; ++i) rm[i] = fminf(rm[i], __shfl_xor(rm[i], m, 32));
    }
    if (laneN == 0) {
        float s = 0.0f;
        #pragma unroll
        for (int i = 0; i < 8; ++i) s += sqrtf(rm[i]);
        atomicAdd(&rowacc, s);
    }
    __syncthreads();
    if (tid == 0) atomicAdd(out, rowacc * (REG1C / (float)N_ROWS));
    for (int c = tid; c < NPROTO; c += 256) atomicMin(&colmin_g[c], colmin_s[c]);
}

// ---------------------------------------------------------------------------
__global__ void col_reduce(const unsigned* __restrict__ colmin, float* __restrict__ out) {
    __shared__ float red[256];
    int t = threadIdx.x;
    float s = sqrtf(__uint_as_float(colmin[t])) +
              sqrtf(__uint_as_float(colmin[t + 256]));
    red[t] = s;
    __syncthreads();
    for (int off = 128; off > 0; off >>= 1) {
        if (t < off) red[t] += red[t + off];
        __syncthreads();
    }
    if (t == 0) atomicAdd(out, red[0] * (REG2C / (float)NPROTO));
}

// ---------------------------------------------------------------------------
extern "C" void kernel_launch(void* const* d_in, const int* in_sizes, int n_in,
                              void* d_out, int out_size, void* d_ws, size_t ws_size,
                              hipStream_t stream) {
    const float* z     = (const float*)d_in[0];
    const float* proto = (const float*)d_in[1];
    float* out = (float*)d_out;

    char* ws = (char*)d_ws;
    unsigned short* ph = (unsigned short*)(ws);                 // 512*256*2 = 256 KB
    unsigned short* pl = (unsigned short*)(ws + 262144);        // 256 KB
    float*    p2     = (float*)(ws + 524288);                   // 2 KB
    unsigned* colmin = (unsigned*)(ws + 526336);                // 2 KB

    init_kernel<<<1, 512, 0, stream>>>(colmin, out);
    proto_prep<<<2, 256, 0, stream>>>(proto, ph, pl, p2);
    proto_dist_kernel<<<N_ROWS / 128, 256, 0, stream>>>(z, ph, pl, p2, colmin, out);
    col_reduce<<<1, 256, 0, stream>>>(colmin, out);
}
